// TTTNeuralCDE_6133213298759
// MI455X (gfx1250) — matrix operand order repaired and checked by reading, not compiled, not run on hardware
//
#include <hip/hip_runtime.h>
#include <math.h>

// Problem constants (from the reference)
#define B_    256
#define T_    512
#define CIN_  33
#define CX_   32
#define H_    256
#define H2_   128
#define NTRT_ 4
#define NCOL2 (H_ * CIN_)   // 8448
#define NT2   (NCOL2 / 16)  // 528 column tiles; 33 per wave with 16 waves
#define BN_EPS 1e-5f

#define SCAN_THREADS 512    // 16 waves per workgroup

typedef __attribute__((ext_vector_type(16))) _Float16 v16h;
typedef __attribute__((ext_vector_type(8)))  float    v8f;
typedef __attribute__((ext_vector_type(4)))  unsigned int v4u;
typedef __attribute__((ext_vector_type(8)))  int      v8i;
typedef __attribute__((ext_vector_type(4)))  int      v4i;

#if defined(__gfx1250__) && __has_builtin(__builtin_amdgcn_tensor_load_to_lds)
#define USE_TDM 1
#else
#define USE_TDM 0
#endif

// ---------------------------------------------------------------------------
// WMMA fragment helpers (CDNA5 16-bit layouts, wave32; see CDNA5 ISA)
// A (16x32 f16): lane reads one contiguous 16-half chunk: [koff+hi*16,+16)
// B (32x16 f16, stored transposed [N][K]): lane reads one contiguous 32B chunk
// C/D (16x16 f32): lane -> col=lane&15; VGPR r -> row = r + 8*(lane>=16)
// ---------------------------------------------------------------------------
__device__ __forceinline__ v16h frag_a16(const _Float16* base, int ld, int lane, int koff) {
  const _Float16* p = base + (lane & 15) * ld + koff + ((lane >> 4) * 16);
  v16h a;
#pragma unroll
  for (int e = 0; e < 8; ++e) a[e] = p[e];
#pragma unroll
  for (int e = 0; e < 8; ++e) a[8 + e] = p[8 + e];
  return a;
}

__device__ __forceinline__ v16h frag_b16(const _Float16* baseT, int ld, int lane, int koff) {
  const _Float16* p = baseT + (lane & 15) * ld + koff + ((lane >> 4) * 16);
  v16h b;
#pragma unroll
  for (int e = 0; e < 16; ++e) b[e] = p[e];
  return b;
}

__device__ __forceinline__ v8f wmma16(v16h a, v16h b, v8f c) {
  return __builtin_amdgcn_wmma_f32_16x16x32_f16(false, a, false, b, (short)0, c, false, false);
}

#if USE_TDM
// ---------------------------------------------------------------------------
// TDM: async-load a [16 x 66] f32 tile (rows = batches, cols = X[t]++X[t+1])
// from global into LDS. D# per CDNA5 ISA:
//   g0: count=1 | lds_addr | global_addr[56:0] | type=2
//   g1: data_size=4B, tensor_dim0=66, tensor_dim1=16, tile_dim0=66,
//       tile_dim1=16, tensor_dim0_stride=T_*CIN_
// This toolchain exposes the 6-arg builtin:
//   (uint32x4 g0, int32x8 g1, int32x4 g2, int32x4 g3, int32x8, i32 cpol)
// groups 2/3 (and the trailing vector) are zero: 2-D tile, no dims 2..4.
// ---------------------------------------------------------------------------
__device__ __forceinline__ void tdm_fetch_xtile(const float* gsrc, void* ldsdst) {
  unsigned long long ga = (unsigned long long)(size_t)gsrc;
  v4u g0;
  g0[0] = 1u;                                                 // count=1, user mode
  g0[1] = (unsigned)(size_t)ldsdst;                           // LDS byte address
  g0[2] = (unsigned)(ga & 0xffffffffu);                       // global_addr lo
  g0[3] = (unsigned)((ga >> 32) & 0x01ffffffu) | 0x80000000u; // addr hi | type=2
  v8i g1;
  g1[0] = (int)(2u << 16);                 // wg_mask=0, data_size=2 (4 bytes)
  g1[1] = (int)(66u << 16);                // tensor_dim0 = 66 (lo16)
  g1[2] = (int)(16u << 16);                // tensor_dim0 hi=0 | tensor_dim1=16 lo
  g1[3] = (int)(66u << 16);                // tensor_dim1 hi=0 | tile_dim0=66
  g1[4] = (int)16u;                        // tile_dim1=16, tile_dim2=0
  g1[5] = (int)(T_ * CIN_);                // tensor_dim0_stride = 16896 (lo32)
  g1[6] = 0;                               // stride hi | tensor_dim1_stride lo
  g1[7] = 0;
  v4i gz4 = {0, 0, 0, 0};
  v8i gz8 = {0, 0, 0, 0, 0, 0, 0, 0};
  __builtin_amdgcn_tensor_load_to_lds(g0, g1, gz4, gz4, gz8, 0);
}
#endif

// ---------------------------------------------------------------------------
// Weight prep: transpose + convert to f16 so B-fragments are contiguous [N][K]
// ---------------------------------------------------------------------------
__global__ __launch_bounds__(256) void prep_weights(
    const float* __restrict__ l1W,  // [256 in][128 out]
    const float* __restrict__ l2W,  // [128][8448]
    const float* __restrict__ aW1,  // [256][128]
    _Float16* __restrict__ l1hT,    // [128][256]
    _Float16* __restrict__ l2hT,    // [8448][128]
    _Float16* __restrict__ attT)    // [128][256]
{
  int i = blockIdx.x * blockDim.x + threadIdx.x;
  int stride = gridDim.x * blockDim.x;
  for (int idx = i; idx < H2_ * H_; idx += stride) {
    int n = idx / H_, k = idx % H_;
    l1hT[n * H_ + k] = (_Float16)l1W[k * H2_ + n];
    attT[n * H_ + k] = (_Float16)aW1[k * H2_ + n];
  }
  for (int idx = i; idx < NCOL2 * H2_; idx += stride) {
    int n = idx / H2_, k = idx % H2_;
    l2hT[n * H2_ + k] = (_Float16)l2W[k * NCOL2 + n];
  }
}

// ---------------------------------------------------------------------------
// h0 = X[:,0,1:] @ embed_W + embed_b
// ---------------------------------------------------------------------------
__global__ __launch_bounds__(256) void embed_h0(
    const float* __restrict__ X, const float* __restrict__ eW,
    const float* __restrict__ eb, float* __restrict__ h0)
{
  int b = blockIdx.x, h = threadIdx.x;
  float acc = eb[h];
#pragma unroll 4
  for (int c = 0; c < CX_; ++c)
    acc += X[(size_t)b * T_ * CIN_ + 1 + c] * eW[c * H_ + h];
  h0[(size_t)b * H_ + h] = acc;
}

// ---------------------------------------------------------------------------
// Batch-norm over batch dim + ReLU: one block per feature h, 256 threads = B
// ---------------------------------------------------------------------------
__global__ __launch_bounds__(256) void bn_relu(
    const float* __restrict__ h0, const float* __restrict__ gamma,
    const float* __restrict__ beta, float* __restrict__ z0)
{
  __shared__ float red[256];
  int h = blockIdx.x, b = threadIdx.x;
  float v = h0[(size_t)b * H_ + h];
  red[b] = v; __syncthreads();
  for (int o = 128; o > 0; o >>= 1) { if (b < o) red[b] += red[b + o]; __syncthreads(); }
  float mu = red[0] * (1.0f / B_); __syncthreads();
  float d = v - mu;
  red[b] = d * d; __syncthreads();
  for (int o = 128; o > 0; o >>= 1) { if (b < o) red[b] += red[b + o]; __syncthreads(); }
  float var = red[0] * (1.0f / B_);
  float zn = d * rsqrtf(var + BN_EPS) * gamma[h] + beta[h];
  z0[(size_t)b * H_ + h] = zn > 0.f ? zn : 0.f;
}

// ---------------------------------------------------------------------------
// Persistent RK4 scan. One workgroup per 16-batch tile; z lives in LDS for all
// 511 steps. 16 waves: stage-A WMMA (waves 0-7), stage-B streams L2-resident
// l2_W tiles via WMMA (33 tiles/wave) fusing the einsum with ds_add_f32.
// dX tiles are double-buffered into LDS by the Tensor Data Mover one step
// ahead (TENSORcnt-pipelined), so no global-load latency sits on the chain.
// ---------------------------------------------------------------------------
__global__ __launch_bounds__(SCAN_THREADS) void scan_rk4(
    const float* __restrict__ X, const float* __restrict__ z0,
    const _Float16* __restrict__ l1hT, const float* __restrict__ l1b,
    const _Float16* __restrict__ l2hT, const float* __restrict__ l2b,
    float* __restrict__ zhat)
{
  __shared__ float    s_z[16][H_];          // 16 KB  current state
  __shared__ float    s_g[16][H_];          // 16 KB  g(z') accumulator (k_s)
  __shared__ _Float16 s_za[16][H_];         //  8 KB  f16 A operand (z + a*k)
  __shared__ _Float16 s_h[16][H2_];         //  4 KB  f16 A operand (relu hidden)
  __shared__ float    s_dx[16][CIN_ + 3];   // ~2 KB  dX tile (padded)
#if USE_TDM
  __shared__ float    s_xs[2][16][2 * CIN_]; // 8.25 KB TDM staging, double-buffered
#endif

  const int tid  = threadIdx.x;
  const int lane = tid & 31;
  const int wv   = tid >> 5;           // 16 waves
  const int b0   = blockIdx.x * 16;    // batch tile
  const int om   = tid >> 5;           // owned row (0..15)
  const int oc0  = (tid & 31) * 8;     // owned 8-col block

  // Load z0 tile, emit z_hat[:,0,:], clear g
#pragma unroll
  for (int j = 0; j < 8; ++j) {
    float v = z0[(size_t)(b0 + om) * H_ + oc0 + j];
    s_z[om][oc0 + j] = v;
    s_g[om][oc0 + j] = 0.f;
    zhat[(size_t)(b0 + om) * T_ * H_ + oc0 + j] = v;
  }
#if USE_TDM
  if (wv == 0)  // kick off the first X tile (rows t=0 and t=1)
    tdm_fetch_xtile(X + (size_t)b0 * T_ * CIN_, &s_xs[0][0][0]);
#endif
  __syncthreads();

  const float alpha[4] = {0.f, 0.5f, 0.5f, 1.f};
  const float wgt[4]   = {1.f, 2.f, 2.f, 1.f};

#pragma unroll 1
  for (int t = 0; t < T_ - 1; ++t) {
#if USE_TDM
    const int cur = t & 1;
    if (wv == 0) {
      if (t < T_ - 2) {
        tdm_fetch_xtile(X + (size_t)b0 * T_ * CIN_ + (size_t)(t + 1) * CIN_,
                        &s_xs[cur ^ 1][0][0]);
        __builtin_amdgcn_s_wait_tensorcnt(1);  // retire tile t, keep t+1 in flight
      } else {
        __builtin_amdgcn_s_wait_tensorcnt(0);
      }
    }
    __syncthreads();  // TDM data for step t visible to all waves
    for (int idx = tid; idx < 16 * CIN_; idx += SCAN_THREADS) {
      int m = idx / CIN_, c = idx % CIN_;
      s_dx[m][c] = s_xs[cur][m][c + CIN_] - s_xs[cur][m][c];
    }
#else
    for (int idx = tid; idx < 16 * CIN_; idx += SCAN_THREADS) {
      int m = idx / CIN_, c = idx % CIN_;
      size_t base = (size_t)(b0 + m) * T_ * CIN_ + (size_t)t * CIN_ + c;
      s_dx[m][c] = X[base + CIN_] - X[base];
    }
#endif
    float racc[8];
#pragma unroll
    for (int j = 0; j < 8; ++j) racc[j] = 0.f;

#pragma unroll 1
    for (int s = 0; s < 4; ++s) {
      const float al = alpha[s];
#pragma unroll
      for (int j = 0; j < 8; ++j)
        s_za[om][oc0 + j] = (_Float16)(s_z[om][oc0 + j] + al * s_g[om][oc0 + j]);
      __syncthreads();
#pragma unroll
      for (int j = 0; j < 8; ++j) s_g[om][oc0 + j] = 0.f;

      // ---- Stage A: h = relu(za @ l1_W + l1_b), [16,256]x[256,128] ----
      // Only 8 col-tiles exist: waves 0-7 run it (uniform per-wave branch,
      // EXEC all-ones inside participating waves as WMMA requires).
      if (wv < 8) {
        const int n0 = wv * 16;
        v8f c = {};
#pragma unroll
        for (int kt = 0; kt < 8; ++kt) {
          v16h a = frag_a16(&s_za[0][0], H_, lane, kt * 32);
          v16h b = frag_b16(l1hT + (size_t)n0 * H_, H_, lane, kt * 32);
          c = wmma16(a, b, c);
        }
        const int n  = n0 + (lane & 15);
        const int mb = (lane >> 4) * 8;
        const float bia = l1b[n];
#pragma unroll
        for (int r = 0; r < 8; ++r) {
          float v = c[r] + bia;
          s_h[mb + r][n] = (_Float16)(v > 0.f ? v : 0.f);
        }
      }
      __syncthreads();

      // ---- Stage B: f = h @ l2_W + l2_b ; g[b,h'] += f[b,n]*dx[b,n%33] ----
#pragma unroll 1
      for (int ct = wv; ct < NT2; ct += 16) {
        const int n0 = ct * 16;
        const _Float16* bbase = l2hT + (size_t)n0 * H2_;
        __builtin_prefetch(bbase + (size_t)16 * 16 * H2_, 0, 0);  // next tile set
        v8f c = {};
#pragma unroll
        for (int kt = 0; kt < 4; ++kt) {
          v16h a = frag_a16(&s_h[0][0], H2_, lane, kt * 32);
          v16h b = frag_b16(bbase, H2_, lane, kt * 32);
          c = wmma16(a, b, c);
        }
        const int n  = n0 + (lane & 15);
        const int hp = n / CIN_;
        const int ci = n % CIN_;
        const int mb = (lane >> 4) * 8;
        const float bia = l2b[n];
#pragma unroll
        for (int r = 0; r < 8; ++r) {
          const int m = mb + r;
          atomicAdd(&s_g[m][hp], (c[r] + bia) * s_dx[m][ci]);  // ds_add_f32
        }
      }
      __syncthreads();

      const float wg = wgt[s];
#pragma unroll
      for (int j = 0; j < 8; ++j) racc[j] += wg * s_g[om][oc0 + j];
    }

    // z += (k1 + 2k2 + 2k3 + k4)/6 ; emit z_hat[:,t+1,:]
#pragma unroll
    for (int j = 0; j < 8; ++j) {
      float zn = s_z[om][oc0 + j] + racc[j] * (1.f / 6.f);
      s_z[om][oc0 + j] = zn;
      zhat[(size_t)(b0 + om) * T_ * H_ + (size_t)(t + 1) * H_ + oc0 + j] = zn;
    }
    // owner-disjoint writes; next iteration's barriers cover dx/za hazards
  }
}

// ---------------------------------------------------------------------------
// Attention scores: tanh(z_hat @ att_W1 + b1) @ att_W2 + b2, WMMA over 16-row
// tiles of the flattened (b,t) axis. 8192 blocks x 8 waves.
// ---------------------------------------------------------------------------
__global__ __launch_bounds__(256) void attn_scores(
    const float* __restrict__ zhat, const _Float16* __restrict__ attT,
    const float* __restrict__ ab1, const float* __restrict__ aW2,
    const float* __restrict__ ab2, float* __restrict__ scores)
{
  __shared__ _Float16 s_za[16][H_];
  __shared__ float    s_part[16];
  const int tid = threadIdx.x, lane = tid & 31, wv = tid >> 5;
  const size_t r0 = (size_t)blockIdx.x * 16;
  const int om = tid >> 4, oc0 = (tid & 15) * 16;
#pragma unroll
  for (int j = 0; j < 16; ++j)
    s_za[om][oc0 + j] = (_Float16)zhat[(r0 + om) * H_ + oc0 + j];
  if (tid < 16) s_part[tid] = 0.f;
  __syncthreads();

  const int n0 = wv * 16;
  v8f c = {};
#pragma unroll
  for (int kt = 0; kt < 8; ++kt) {
    v16h a = frag_a16(&s_za[0][0], H_, lane, kt * 32);
    v16h b = frag_b16(attT + (size_t)n0 * H_, H_, lane, kt * 32);
    c = wmma16(a, b, c);
  }
  const int n  = n0 + (lane & 15);
  const int mb = (lane >> 4) * 8;
  const float b1 = ab1[n], w2 = aW2[n];
#pragma unroll
  for (int r = 0; r < 8; ++r) {
    float v = tanhf(c[r] + b1);
    atomicAdd(&s_part[mb + r], v * w2);
  }
  __syncthreads();
  if (tid < 16) scores[r0 + tid] = s_part[tid] + ab2[0];
}

// ---------------------------------------------------------------------------
// Softmax over T + weighted pooling. One block per batch row.
// ---------------------------------------------------------------------------
__global__ __launch_bounds__(256) void softmax_pool(
    const float* __restrict__ scores, const float* __restrict__ zhat,
    float* __restrict__ zpool)
{
  __shared__ float red[256];
  __shared__ float wts[T_];
  const int b = blockIdx.x, tid = threadIdx.x;
  const float* sc = scores + (size_t)b * T_;
  red[tid] = fmaxf(sc[tid], sc[tid + 256]); __syncthreads();
  for (int o = 128; o > 0; o >>= 1) { if (tid < o) red[tid] = fmaxf(red[tid], red[tid + o]); __syncthreads(); }
  const float mx = red[0]; __syncthreads();
  float e0 = expf(sc[tid] - mx), e1 = expf(sc[tid + 256] - mx);
  wts[tid] = e0; wts[tid + 256] = e1;
  red[tid] = e0 + e1; __syncthreads();
  for (int o = 128; o > 0; o >>= 1) { if (tid < o) red[tid] += red[tid + o]; __syncthreads(); }
  const float inv = 1.f / red[0]; __syncthreads();
  wts[tid] *= inv; wts[tid + 256] *= inv;
  __syncthreads();
  float acc = 0.f;
  const float* zb = zhat + (size_t)b * T_ * H_ + tid;   // tid == feature h
#pragma unroll 4
  for (int t = 0; t < T_; ++t) acc += wts[t] * zb[(size_t)t * H_];
  zpool[(size_t)b * H_ + tid] = acc;
}

// ---------------------------------------------------------------------------
// Output heads: pred_y, pred_a, softmax(pred_a). One block per batch row.
// ---------------------------------------------------------------------------
__global__ __launch_bounds__(128) void heads(
    const float* __restrict__ zpool,
    const float* __restrict__ oW1, const float* __restrict__ ob1,
    const float* __restrict__ oW2, const float* __restrict__ ob2,
    const float* __restrict__ tW1, const float* __restrict__ tb1,
    const float* __restrict__ tW2, const float* __restrict__ tb2,
    float* __restrict__ out)
{
  __shared__ float s_zp[H_];
  __shared__ float s_hy[H2_], s_ha[H2_];
  __shared__ float s_pa[NTRT_];
  const int b = blockIdx.x, j = threadIdx.x;
  s_zp[j]       = zpool[(size_t)b * H_ + j];
  s_zp[j + 128] = zpool[(size_t)b * H_ + j + 128];
  __syncthreads();
  float dy = 0.f, da = 0.f;
#pragma unroll 4
  for (int k = 0; k < H_; ++k) {
    float z = s_zp[k];
    dy += z * oW1[k * H2_ + j];
    da += z * tW1[k * H2_ + j];
  }
  dy += ob1[j]; da += tb1[j];
  s_hy[j] = dy > 0.f ? dy : 0.f;
  s_ha[j] = da > 0.f ? da : 0.f;
  __syncthreads();
  if (j == 0) {
    float py = 0.f;
    for (int k = 0; k < H2_; ++k) py += s_hy[k] * oW2[k];
    out[b] = py + ob2[0];                              // pred_y @ [0, 256)
  }
  if (j < NTRT_) {
    float pa = 0.f;
    for (int k = 0; k < H2_; ++k) pa += s_ha[k] * tW2[k * NTRT_ + j];
    pa += tb2[j];
    s_pa[j] = pa;
    out[B_ + B_ * NTRT_ + b * NTRT_ + j] = pa;         // pred_a @ [1280, 2304)
  }
  __syncthreads();
  if (j == 0) {
    float mx = s_pa[0];
    for (int k = 1; k < NTRT_; ++k) mx = fmaxf(mx, s_pa[k]);
    float e[NTRT_], sm = 0.f;
    for (int k = 0; k < NTRT_; ++k) { e[k] = expf(s_pa[k] - mx); sm += e[k]; }
    float inv = 1.f / sm;
    for (int k = 0; k < NTRT_; ++k)
      out[B_ + b * NTRT_ + k] = e[k] * inv;            // softmax(pred_a) @ [256, 1280)
  }
}

// ---------------------------------------------------------------------------
extern "C" void kernel_launch(void* const* d_in, const int* in_sizes, int n_in,
                              void* d_out, int out_size, void* d_ws, size_t ws_size,
                              hipStream_t stream) {
  const float* X    = (const float*)d_in[0];   // coeffs_x  [B,T,CIN]
  const float* eW   = (const float*)d_in[1];
  const float* eb   = (const float*)d_in[2];
  const float* gmm  = (const float*)d_in[3];
  const float* bta  = (const float*)d_in[4];
  const float* l1W  = (const float*)d_in[5];
  const float* l1b  = (const float*)d_in[6];
  const float* l2W  = (const float*)d_in[7];
  const float* l2b  = (const float*)d_in[8];
  const float* aW1  = (const float*)d_in[9];
  const float* ab1  = (const float*)d_in[10];
  const float* aW2  = (const float*)d_in[11];
  const float* ab2  = (const float*)d_in[12];
  const float* oW1  = (const float*)d_in[13];
  const float* ob1  = (const float*)d_in[14];
  const float* oW2  = (const float*)d_in[15];
  const float* ob2  = (const float*)d_in[16];
  const float* tW1  = (const float*)d_in[17];
  const float* tb1  = (const float*)d_in[18];
  const float* tW2  = (const float*)d_in[19];
  const float* tb2  = (const float*)d_in[20];

  // Workspace layout
  char* p = (char*)d_ws;
  float*    zhat   = (float*)p;     p += (size_t)B_ * T_ * H_ * sizeof(float);   // 134 MB
  _Float16* l1hT   = (_Float16*)p;  p += (size_t)H2_ * H_ * sizeof(_Float16);
  _Float16* l2hT   = (_Float16*)p;  p += (size_t)NCOL2 * H2_ * sizeof(_Float16); // 2.1 MB
  _Float16* attT   = (_Float16*)p;  p += (size_t)H2_ * H_ * sizeof(_Float16);
  float*    h0     = (float*)p;     p += (size_t)B_ * H_ * sizeof(float);
  float*    z0     = (float*)p;     p += (size_t)B_ * H_ * sizeof(float);
  float*    scores = (float*)p;     p += (size_t)B_ * T_ * sizeof(float);

  float* out   = (float*)d_out;
  float* zpool = out + B_ + 2 * B_ * NTRT_;   // z_pool @ [2304, 67840)

  prep_weights<<<512, 256, 0, stream>>>(l1W, l2W, aW1, l1hT, l2hT, attT);
  embed_h0    <<<B_, H_,  0, stream>>>(X, eW, eb, h0);
  bn_relu     <<<H_, B_,  0, stream>>>(h0, gmm, bta, z0);
  scan_rk4    <<<B_ / 16, SCAN_THREADS, 0, stream>>>(X, z0, l1hT, l1b, l2hT, l2b, zhat);
  attn_scores <<<(B_ * T_) / 16, 256, 0, stream>>>(zhat, attT, ab1, aW2, ab2, scores);
  softmax_pool<<<B_, 256, 0, stream>>>(scores, zhat, zpool);
  heads       <<<B_, 128, 0, stream>>>(zpool, oW1, ob1, oW2, ob2, tW1, tb1, tW2, tb2, out);
}
